// BertNer_60730837565455
// MI455X (gfx1250) — matrix operand order repaired and checked by reading, not compiled, not run on hardware
//
#include <hip/hip_runtime.h>
#include <hip/hip_bf16.h>
#include <stdint.h>

// ---------------- problem constants ----------------
constexpr int kB  = 16;
constexpr int kS  = 256;
constexpr int kH  = 768;
constexpr int kL  = 12;
constexpr int kA  = 12;
constexpr int kDH = 64;      // kH / kA
constexpr int kFF = 3072;    // 4*kH
constexpr int kNL = 9;
constexpr int kBS = kB * kS; // 4096 tokens

typedef __attribute__((ext_vector_type(16))) _Float16 v16h;
typedef __attribute__((ext_vector_type(8)))  _Float16 v8h;
typedef __attribute__((ext_vector_type(2)))  _Float16 v2h;
typedef __attribute__((ext_vector_type(8)))  float    v8f;

union HFrag { v16h v; v8h h[2]; };

// ---------------- device helpers ----------------
__device__ __forceinline__ float gelu_tanh(float x) {
    float x3 = x * x * x;
    float t  = tanhf(0.7978845608028654f * (x + 0.044715f * x3));
    return 0.5f * x * (1.0f + t);
}

// issue one 16-byte async global->LDS copy (ASYNCcnt-tracked TDM-lite path)
__device__ __forceinline__ void async_copy_b128(const float* gsrc, const float* lds_dst) {
    asm volatile("global_load_async_to_lds_b128 %0, %1, off"
                 :: "v"((unsigned)(uintptr_t)lds_dst),
                    "v"((unsigned long long)(uintptr_t)gsrc)
                 : "memory");
}
__device__ __forceinline__ void wait_async0() {
    asm volatile("s_wait_asynccnt 0x0" ::: "memory");
}

// convert 8 f32 (two float4) into elements [base, base+8) of a v16h
__device__ __forceinline__ void cvt8(v16h& d, int base, float4 a, float4 b) {
    d[base + 0] = (_Float16)a.x; d[base + 1] = (_Float16)a.y;
    d[base + 2] = (_Float16)a.z; d[base + 3] = (_Float16)a.w;
    d[base + 4] = (_Float16)b.x; d[base + 5] = (_Float16)b.y;
    d[base + 6] = (_Float16)b.z; d[base + 7] = (_Float16)b.w;
}

// =====================================================================
// LDS-staged batched WMMA GEMM:  C = scale * (A @ B) + bias[n] (opt GELU)
//   A: f32, element (m,k) at Abase[m*lda + k]
//   B: f32, element (k,n) at Bbase[k*ldb + n]  (TRANSB: Bbase[n*ldb + k])
// 256 threads = 8 waves; block computes a 128x64 macro-tile, BK=32.
// A tile (128x32 f32, 16 KB) staged with global_load_async_to_lds_b128;
// B tile likewise when TRANSB (k-contiguous), else batched sync f32->f16.
// Each wave: 2x2 v_wmma_f32_16x16x32_f16 on a 32x32 sub-tile; f32->f16
// fragment conversion (v_cvt_pk_f16_f32) co-executes with the XDL WMMAs.
// All shapes divide BM/BN/BK exactly -> no divergence, EXEC always full.
// =====================================================================
template<int TRANSB, int GELU>
__global__ void __launch_bounds__(256)
wmma_gemm_tiled(const float* __restrict__ A, int lda, long sAb, long sAh,
                const float* __restrict__ Bm, int ldb, long sBb, long sBh,
                const float* __restrict__ bias, long sBiasB,
                float* __restrict__ C, int ldc, long sCb, long sCh,
                int M, int N, int K, float scale)
{
    constexpr int BM = 128, BN = 64, BK = 32;
    __shared__ float As32[BM * BK];            // 16 KB f32 A tile, [m][k]
    __shared__ float Bbuf[BN * BK];            // 8 KB: f32 [n][k] (TRANSB) or f16 [n][k]
    _Float16* Bsh = (_Float16*)Bbuf;

    const int tid    = threadIdx.x;
    const int lane   = tid & 31;
    const int wave   = tid >> 5;
    const int tilesN = N / BN;
    const int tm0 = (blockIdx.x / tilesN) * BM;
    const int tn0 = (blockIdx.x % tilesN) * BN;

    const float* Ab = A  + (long)blockIdx.y * sAb + (long)blockIdx.z * sAh;
    const float* Bb = Bm + (long)blockIdx.y * sBb + (long)blockIdx.z * sBh;
    float*       Cb = C  + (long)blockIdx.y * sCb + (long)blockIdx.z * sCh;

    // 4x2 wave grid; each wave owns a 32x32 sub-tile (2x2 WMMA tiles)
    const int wm0  = (wave >> 1) * 32;
    const int wn0  = (wave & 1) * 32;
    const int mr   = lane & 15;
    const int half = lane >> 4;

    v8f acc[2][2] = {};

    for (int k0 = 0; k0 < K; k0 += BK) {
        // ---- stage A: 128x32 f32 via async copies (1024 x 16B, 4/thread) ----
#pragma unroll
        for (int it = 0; it < 4; ++it) {
            const int c    = tid + it * 256;
            const int row  = c >> 3;
            const int koff = (c & 7) * 4;
            const float* gp = Ab + (long)(tm0 + row) * lda + k0 + koff;
            async_copy_b128(gp, &As32[row * BK + koff]);
            __builtin_prefetch(gp + BK, 0, 1);           // speculative: safe past end
        }
        // ---- stage B ----
        if (TRANSB) {
            // k-contiguous rows: async copies (512 x 16B, 2/thread), f32 [n][k]
#pragma unroll
            for (int it = 0; it < 2; ++it) {
                const int c    = tid + it * 256;
                const int n    = c >> 3;
                const int koff = (c & 7) * 4;
                const float* gp = Bb + (long)(tn0 + n) * ldb + k0 + koff;
                async_copy_b128(gp, &Bbuf[n * BK + koff]);
                __builtin_prefetch(gp + BK, 0, 1);
            }
        } else {
            // strided columns: batch scalar loads, then convert+store f16 [n][k]
            float e0[4], e1[4];
#pragma unroll
            for (int it = 0; it < 4; ++it) {
                const int p  = tid + it * 256;
                const int n  = p >> 4;
                const int kp = (p & 15) * 2;
                e0[it] = Bb[(long)(k0 + kp)     * ldb + tn0 + n];
                e1[it] = Bb[(long)(k0 + kp + 1) * ldb + tn0 + n];
            }
#pragma unroll
            for (int it = 0; it < 4; ++it) {
                const int p  = tid + it * 256;
                const int n  = p >> 4;
                const int kp = (p & 15) * 2;
                v2h h; h[0] = (_Float16)e0[it]; h[1] = (_Float16)e1[it];
                *(v2h*)(&Bsh[n * BK + kp]) = h;
            }
        }
        wait_async0();
        __syncthreads();

        // ---- fragment loads (aligned ds_load_b128) + f32->f16 convert ----
        // A 16x32 layout: lanes 0-15 hold k=0..15, lanes 16-31 hold k=16..31
        // B 32x16 layout: lanes 0-15 hold k=0..15, lanes 16-31 hold k=16..31
        HFrag a[2], b[2];
#pragma unroll
        for (int i = 0; i < 2; ++i) {
            const float* ar = &As32[(wm0 + i * 16 + mr) * BK];
            float4 q0 = *(const float4*)(ar + half * 16);
            float4 q1 = *(const float4*)(ar + half * 16 + 4);
            float4 q2 = *(const float4*)(ar + half * 16 + 8);
            float4 q3 = *(const float4*)(ar + half * 16 + 12);
            cvt8(a[i].v, 0, q0, q1);
            cvt8(a[i].v, 8, q2, q3);
            if (TRANSB) {
                const float* br = &Bbuf[(wn0 + i * 16 + mr) * BK];
                float4 r0 = *(const float4*)(br + half * 16);
                float4 r1 = *(const float4*)(br + half * 16 + 4);
                float4 r2 = *(const float4*)(br + half * 16 + 8);
                float4 r3 = *(const float4*)(br + half * 16 + 12);
                cvt8(b[i].v, 0, r0, r1);
                cvt8(b[i].v, 8, r2, r3);
            } else {
                const _Float16* br = &Bsh[(wn0 + i * 16 + mr) * BK];
                b[i].h[0] = *(const v8h*)(br + half * 16);
                b[i].h[1] = *(const v8h*)(br + half * 16 + 8);
            }
        }
#pragma unroll
        for (int i = 0; i < 2; ++i)
#pragma unroll
            for (int j = 0; j < 2; ++j)
                acc[i][j] = __builtin_amdgcn_wmma_f32_16x16x32_f16(
                    false, a[i].v, false, b[j].v, (short)0, acc[i][j], false, false);
        __syncthreads();
    }

    // ---- epilogue: C layout VGPR r -> row r (+8 for upper half), n = lane%16 ----
#pragma unroll
    for (int j = 0; j < 2; ++j) {
        const int n = tn0 + wn0 + j * 16 + mr;
        const float bval = bias ? (bias + (long)blockIdx.y * sBiasB)[n] : 0.0f;
#pragma unroll
        for (int i = 0; i < 2; ++i) {
#pragma unroll
            for (int r = 0; r < 8; ++r) {
                const int m = tm0 + wm0 + i * 16 + r + half * 8;
                float v = acc[i][j][r] * scale + bval;
                if (GELU) v = gelu_tanh(v);
                Cb[(long)m * ldc + n] = v;
            }
        }
    }
}

// =====================================================================
// Embedding gather + LayerNorm  (one 256-thread block per token)
// =====================================================================
__global__ void embed_ln_kernel(const int* __restrict__ ids,
                                const int* __restrict__ tids,
                                const float* __restrict__ wemb,
                                const float* __restrict__ pemb,
                                const float* __restrict__ temb,
                                const float* __restrict__ g,
                                const float* __restrict__ b,
                                float* __restrict__ x)
{
    const int tok = blockIdx.x;            // [0, kBS)
    const int ss  = tok % kS;
    const int t   = threadIdx.x;           // 256 threads, 3 elems each
    __shared__ float red[256];

    const int wid = ids[tok];
    const int tyi = tids[tok];
    float vals[3];
    float lsum = 0.0f;
#pragma unroll
    for (int i = 0; i < 3; ++i) {
        const int h = t + i * 256;
        float v = wemb[(long)wid * kH + h] + pemb[ss * kH + h] + temb[tyi * kH + h];
        vals[i] = v; lsum += v;
    }
    red[t] = lsum; __syncthreads();
    for (int s = 128; s > 0; s >>= 1) { if (t < s) red[t] += red[t + s]; __syncthreads(); }
    const float mean = red[0] / (float)kH; __syncthreads();

    float lvar = 0.0f;
#pragma unroll
    for (int i = 0; i < 3; ++i) { float d = vals[i] - mean; lvar += d * d; }
    red[t] = lvar; __syncthreads();
    for (int s = 128; s > 0; s >>= 1) { if (t < s) red[t] += red[t + s]; __syncthreads(); }
    const float rstd = rsqrtf(red[0] / (float)kH + 1e-12f);

#pragma unroll
    for (int i = 0; i < 3; ++i) {
        const int h = t + i * 256;
        x[(long)tok * kH + h] = (vals[i] - mean) * rstd * g[h] + b[h];
    }
}

// =====================================================================
// x = LayerNorm(x + delta)  (one block per token)
// =====================================================================
__global__ void add_ln_kernel(float* __restrict__ x,
                              const float* __restrict__ delta,
                              const float* __restrict__ g,
                              const float* __restrict__ b)
{
    const int tok = blockIdx.x;
    const int t   = threadIdx.x;
    __shared__ float red[256];

    float vals[3];
    float lsum = 0.0f;
#pragma unroll
    for (int i = 0; i < 3; ++i) {
        const int h = t + i * 256;
        float v = x[(long)tok * kH + h] + delta[(long)tok * kH + h];
        vals[i] = v; lsum += v;
    }
    red[t] = lsum; __syncthreads();
    for (int s = 128; s > 0; s >>= 1) { if (t < s) red[t] += red[t + s]; __syncthreads(); }
    const float mean = red[0] / (float)kH; __syncthreads();

    float lvar = 0.0f;
#pragma unroll
    for (int i = 0; i < 3; ++i) { float d = vals[i] - mean; lvar += d * d; }
    red[t] = lvar; __syncthreads();
    for (int s = 128; s > 0; s >>= 1) { if (t < s) red[t] += red[t + s]; __syncthreads(); }
    const float rstd = rsqrtf(red[0] / (float)kH + 1e-12f);

#pragma unroll
    for (int i = 0; i < 3; ++i) {
        const int h = t + i * 256;
        x[(long)tok * kH + h] = (vals[i] - mean) * rstd * g[h] + b[h];
    }
}

// =====================================================================
// maskbias[b,t] = (1 - input_mask[b,t]) * -10000
// =====================================================================
__global__ void maskbias_kernel(const int* __restrict__ mask, float* __restrict__ mb)
{
    const int i = blockIdx.x * 256 + threadIdx.x;
    mb[i] = (1.0f - (float)mask[i]) * -10000.0f;
}

// =====================================================================
// Row softmax over length-256 rows (one block per row)
// =====================================================================
__global__ void softmax_kernel(float* __restrict__ p)
{
    __shared__ float red[256];
    float* row = p + (long)blockIdx.x * kS;
    const int t = threadIdx.x;
    const float v = row[t];
    red[t] = v; __syncthreads();
    for (int s = 128; s > 0; s >>= 1) { if (t < s) red[t] = fmaxf(red[t], red[t + s]); __syncthreads(); }
    const float mx = red[0]; __syncthreads();
    const float e = __expf(v - mx);
    red[t] = e; __syncthreads();
    for (int s = 128; s > 0; s >>= 1) { if (t < s) red[t] += red[t + s]; __syncthreads(); }
    row[t] = e / red[0];
}

// =====================================================================
// Valid-mask token compaction (stable): valid rows to front, rest zero
// one block of 256 per batch
// =====================================================================
__global__ void compact_kernel(const float* __restrict__ x,
                               const int* __restrict__ valid,
                               float* __restrict__ gout)
{
    const int bb = blockIdx.x;
    const int t  = threadIdx.x;            // == source position s
    __shared__ int scan[256];
    const int v = valid[bb * kS + t];
    scan[t] = v; __syncthreads();
    for (int off = 1; off < 256; off <<= 1) {
        int add = (t >= off) ? scan[t - off] : 0;
        __syncthreads();
        scan[t] += add;
        __syncthreads();
    }
    const int excl = scan[t] - v;          // exclusive prefix = destination slot

    // zero the whole output slab for this batch (each thread zeroes one row)
    float* zrow = gout + ((long)bb * kS + t) * kH;
    for (int h = 0; h < kH; ++h) zrow[h] = 0.0f;
    __syncthreads();

    if (v) {
        const float* src = x    + ((long)bb * kS + t)    * kH;
        float*       dst = gout + ((long)bb * kS + excl) * kH;
        for (int h = 0; h < kH; ++h) dst[h] = src[h];
    }
}

// =====================================================================
// Classifier head: out = softmax(row @ cls_W + cls_b), 9 labels
// one 32-thread wave per row
// =====================================================================
__global__ void classify_kernel(const float* __restrict__ gin,
                                const float* __restrict__ W,
                                const float* __restrict__ bcls,
                                float* __restrict__ out)
{
    const int row = blockIdx.x;            // [0, kBS)
    const int t   = threadIdx.x;           // 32
    __shared__ float logits[kNL];
    if (t < kNL) {
        const float* xr = gin + (long)row * kH;
        float acc = bcls[t];
        for (int h = 0; h < kH; ++h) acc += xr[h] * W[h * kNL + t];
        logits[t] = acc;
    }
    __syncthreads();
    if (t == 0) {
        float mx = -1e30f;
        for (int j = 0; j < kNL; ++j) mx = fmaxf(mx, logits[j]);
        float e[kNL]; float s = 0.0f;
        for (int j = 0; j < kNL; ++j) { e[j] = __expf(logits[j] - mx); s += e[j]; }
        const float inv = 1.0f / s;
        for (int j = 0; j < kNL; ++j) out[(long)row * kNL + j] = e[j] * inv;
    }
}

// ---------------- host-side GEMM launcher (template dispatch) ----------------
static inline void launch_gemm(hipStream_t st,
                               const float* A, int lda, long sAb, long sAh,
                               const float* Bm, int ldb, long sBb, long sBh, int transB,
                               const float* bias, long sBiasB,
                               float* C, int ldc, long sCb, long sCh,
                               int M, int N, int K, float scale, int gelu,
                               int by, int bz)
{
    dim3 grid((M / 128) * (N / 64), by, bz);
    if (transB) {
        if (gelu) wmma_gemm_tiled<1, 1><<<grid, 256, 0, st>>>(A, lda, sAb, sAh, Bm, ldb, sBb, sBh,
                                                              bias, sBiasB, C, ldc, sCb, sCh, M, N, K, scale);
        else      wmma_gemm_tiled<1, 0><<<grid, 256, 0, st>>>(A, lda, sAb, sAh, Bm, ldb, sBb, sBh,
                                                              bias, sBiasB, C, ldc, sCb, sCh, M, N, K, scale);
    } else {
        if (gelu) wmma_gemm_tiled<0, 1><<<grid, 256, 0, st>>>(A, lda, sAb, sAh, Bm, ldb, sBb, sBh,
                                                              bias, sBiasB, C, ldc, sCb, sCh, M, N, K, scale);
        else      wmma_gemm_tiled<0, 0><<<grid, 256, 0, st>>>(A, lda, sAb, sAh, Bm, ldb, sBb, sBh,
                                                              bias, sBiasB, C, ldc, sCb, sCh, M, N, K, scale);
    }
}

extern "C" void kernel_launch(void* const* d_in, const int* in_sizes, int n_in,
                              void* d_out, int out_size, void* d_ws, size_t ws_size,
                              hipStream_t stream)
{
    const int*   ids   = (const int*)d_in[0];
    const int*   imask = (const int*)d_in[1];
    const int*   tids  = (const int*)d_in[2];
    const int*   vmask = (const int*)d_in[3];
    const float* wemb  = (const float*)d_in[4];
    const float* pemb  = (const float*)d_in[5];
    const float* temb  = (const float*)d_in[6];
    const float* elng  = (const float*)d_in[7];
    const float* elnb  = (const float*)d_in[8];
    const float* Wq    = (const float*)d_in[9];
    const float* bq    = (const float*)d_in[10];
    const float* Wk    = (const float*)d_in[11];
    const float* bk    = (const float*)d_in[12];
    const float* Wv    = (const float*)d_in[13];
    const float* bv    = (const float*)d_in[14];
    const float* Wo    = (const float*)d_in[15];
    const float* bo    = (const float*)d_in[16];
    const float* alng  = (const float*)d_in[17];
    const float* alnb  = (const float*)d_in[18];
    const float* W1    = (const float*)d_in[19];
    const float* b1    = (const float*)d_in[20];
    const float* W2    = (const float*)d_in[21];
    const float* b2    = (const float*)d_in[22];
    const float* flng  = (const float*)d_in[23];
    const float* flnb  = (const float*)d_in[24];
    const float* clsW  = (const float*)d_in[25];
    const float* clsb  = (const float*)d_in[26];
    float* out = (float*)d_out;

    // ---- workspace carve-up (f32 elements) ----
    float* ws = (float*)d_ws;
    size_t off = 0;
    float* x    = ws + off; off += (size_t)kBS * kH;          // activations
    float* q_   = ws + off; off += (size_t)kBS * kH;          // Q (later: gathered)
    float* k_   = ws + off; off += (size_t)kBS * kH;          // K (later: tmp)
    float* v_   = ws + off; off += (size_t)kBS * kH;          // V
    float* ctx  = ws + off; off += (size_t)kBS * kH;          // attention context
    float* big  = ws + off; off += (size_t)kB * kA * kS * kS; // scores / FFN hidden
    float* mb   = ws + off; off += (size_t)kB * kS;           // mask bias rows

    // ---- embeddings + mask bias ----
    maskbias_kernel<<<dim3(kB * kS / 256), 256, 0, stream>>>(imask, mb);
    embed_ln_kernel<<<dim3(kBS), 256, 0, stream>>>(ids, tids, wemb, pemb, temb,
                                                   elng, elnb, x);

    // ---- 12 transformer layers ----
    for (int l = 0; l < kL; ++l) {
        const float* wq = Wq + (size_t)l * kH * kH;
        const float* wk = Wk + (size_t)l * kH * kH;
        const float* wv = Wv + (size_t)l * kH * kH;
        const float* wo = Wo + (size_t)l * kH * kH;
        const float* w1 = W1 + (size_t)l * kH * kFF;
        const float* w2 = W2 + (size_t)l * kFF * kH;

        // Q/K/V projections: [4096,768] @ [768,768]
        launch_gemm(stream, x, kH, 0, 0, wq, kH, 0, 0, 0, bq + (size_t)l * kH, 0,
                    q_, kH, 0, 0, kBS, kH, kH, 1.0f, 0, 1, 1);
        launch_gemm(stream, x, kH, 0, 0, wk, kH, 0, 0, 0, bk + (size_t)l * kH, 0,
                    k_, kH, 0, 0, kBS, kH, kH, 1.0f, 0, 1, 1);
        launch_gemm(stream, x, kH, 0, 0, wv, kH, 0, 0, 0, bv + (size_t)l * kH, 0,
                    v_, kH, 0, 0, kBS, kH, kH, 1.0f, 0, 1, 1);

        // scores[b,h] = Q Kᵀ / sqrt(DH) + maskbias[b]   (256x256, K=64, batched b×h)
        launch_gemm(stream,
                    q_, kH, (long)kS * kH, kDH,
                    k_, kH, (long)kS * kH, kDH, /*transB=*/1,
                    mb, kS,
                    big, kS, (long)kA * kS * kS, (long)kS * kS,
                    kS, kS, kDH, 0.125f, 0, kB, kA);

        // softmax over all B*A*S rows of length S
        softmax_kernel<<<dim3(kB * kA * kS), 256, 0, stream>>>(big);

        // ctx[b,h] = P V  (256x64, K=256, batched)
        launch_gemm(stream,
                    big, kS, (long)kA * kS * kS, (long)kS * kS,
                    v_, kH, (long)kS * kH, kDH, /*transB=*/0,
                    nullptr, 0,
                    ctx, kH, (long)kS * kH, kDH,
                    kS, kDH, kS, 1.0f, 0, kB, kA);

        // output projection into tmp (reuse k_), then x = LN(x + tmp)
        launch_gemm(stream, ctx, kH, 0, 0, wo, kH, 0, 0, 0, bo + (size_t)l * kH, 0,
                    k_, kH, 0, 0, kBS, kH, kH, 1.0f, 0, 1, 1);
        add_ln_kernel<<<dim3(kBS), 256, 0, stream>>>(x, k_,
                                                     alng + (size_t)l * kH,
                                                     alnb + (size_t)l * kH);

        // FFN: big = gelu(x @ W1 + b1), tmp = big @ W2 + b2, x = LN(x + tmp)
        launch_gemm(stream, x, kH, 0, 0, w1, kFF, 0, 0, 0, b1 + (size_t)l * kFF, 0,
                    big, kFF, 0, 0, kBS, kFF, kH, 1.0f, /*gelu=*/1, 1, 1);
        launch_gemm(stream, big, kFF, 0, 0, w2, kH, 0, 0, 0, b2 + (size_t)l * kH, 0,
                    k_, kH, 0, 0, kBS, kH, kFF, 1.0f, 0, 1, 1);
        add_ln_kernel<<<dim3(kBS), 256, 0, stream>>>(x, k_,
                                                     flng + (size_t)l * kH,
                                                     flnb + (size_t)l * kH);
    }

    // ---- compaction + classifier head ----
    compact_kernel<<<dim3(kB), 256, 0, stream>>>(x, vmask, q_);
    classify_kernel<<<dim3(kBS), 32, 0, stream>>>(q_, clsW, clsb, out);
}
